// VectorQuantizer_32280974197357
// MI455X (gfx1250) — matrix operand order repaired and checked by reading, not compiled, not run on hardware
//
#include <hip/hip_runtime.h>
#include <hip/hip_bf16.h>
#include <stdint.h>

typedef __attribute__((ext_vector_type(16))) __bf16        v16bf;
typedef __attribute__((ext_vector_type(8)))  float         v8f;
typedef __attribute__((ext_vector_type(8)))  unsigned int  v8u;

#define NUM_CODES 1024
#define CDIM      256
#define HW        4096
#define BATCH     32
#define M_TILE    128
#define N_ROWS    (BATCH * HW)        // 131072 query vectors
#define ZQ_NUMEL  (N_ROWS * CDIM)     // 33554432
#define NTILES    (N_ROWS / M_TILE)   // 1024 workgroups

// ---- LDS layout (bytes) ----
#define OFF_AHI   0                   // 64 KB : A hi fragments (frag order)
#define OFF_ALO   65536               // 64 KB : A lo fragments
#define OFF_REDV  131072              // 8 KB  : per-class min values
#define OFF_REDI  139264              // 8 KB  : per-class min indices
#define OFF_IDX   147456              // 512 B : winning index per row
#define OFF_LRED  147968              // 1 KB  : loss reduce scratch
#define LDS_BYTES 148992
// phase-3 z_q tile overlays [0, 128*260*4) = 133120 B (A frags dead by then)
#define ZQ_PITCH  260                 // multiple of 4 -> 16B-aligned async LDS dests

__device__ __forceinline__ unsigned short f32_to_bf16_rn(float x) {
  unsigned u = __float_as_uint(x);
  return (unsigned short)((u + 0x7FFFu + ((u >> 16) & 1u)) >> 16);
}
__device__ __forceinline__ float bf16_bits_to_f32(unsigned short h) {
  return __uint_as_float(((unsigned)h) << 16);
}

// ---------------------------------------------------------------------------
// Kernel 0: quantize codebook to bf16 once, compute ||e_hat||^2 per code.
// ---------------------------------------------------------------------------
__global__ void vq_prep(const float* __restrict__ cb,
                        unsigned short* __restrict__ cb_hi,
                        float* __restrict__ enorm) {
  __shared__ float sred[256];
  const int k = blockIdx.x, c = threadIdx.x;
  float x = cb[k * CDIM + c];
  unsigned short h = f32_to_bf16_rn(x);
  cb_hi[k * CDIM + c] = h;
  float hf = bf16_bits_to_f32(h);
  sred[c] = hf * hf;
  __syncthreads();
  for (int s = 128; s > 0; s >>= 1) {
    if (c < s) sred[c] += sred[c + s];
    __syncthreads();
  }
  if (c == 0) enorm[k] = sred[0];
}

// ---------------------------------------------------------------------------
// Kernel 1: fused distance-GEMM (bf16 WMMA, split-precision, dual acc chains)
//           + argmin + async z_q gather + loss partial. 128 rows per block.
// ---------------------------------------------------------------------------
__global__ void __launch_bounds__(256, 1)
vq_main(const float* __restrict__ z_e, const float* __restrict__ cb_f32,
        const unsigned short* __restrict__ cb_hi, const float* __restrict__ enorm,
        float* __restrict__ out_zq, float* __restrict__ out_idx,
        float* __restrict__ partials) {
  extern __shared__ char smem[];
  unsigned short* ahi  = (unsigned short*)(smem + OFF_AHI);
  unsigned short* alo  = (unsigned short*)(smem + OFF_ALO);
  float*          redv = (float*)(smem + OFF_REDV);
  int*            redi = (int*)(smem + OFF_REDI);
  int*            idxs = (int*)(smem + OFF_IDX);
  float*          lred = (float*)(smem + OFF_LRED);
  float*          zq   = (float*)(smem);   // phase-3 overlay

  const int tid  = threadIdx.x;
  const int tile = blockIdx.x;
  const int b    = tile >> 5;              // 32 tiles per image (HW/M_TILE)
  const int hw0  = (tile & 31) * M_TILE;
  const size_t zbase = (size_t)b * CDIM * HW + hw0;

  // ---- Phase 1: load z tile coalesced, hi/lo split, store in frag order ----
  #pragma unroll 4
  for (int e = tid; e < M_TILE * CDIM; e += 256) {
    int c = e >> 7, j = e & (M_TILE - 1);
    float x = z_e[zbase + (size_t)c * HW + j];
    unsigned short h = f32_to_bf16_rn(x);
    float lo = x - bf16_bits_to_f32(h);
    unsigned short l = f32_to_bf16_rn(lo);
    // 16-bit A-matrix 16x32 mapping, contiguous k: lane half = k/16, elem = k%16
    int w        = j >> 4;
    int cbk      = c >> 5, cc = c & 31;
    int laneHalf = cc >> 4;
    int elem     = cc & 15;
    int lane_s   = (j & 15) | (laneHalf << 4);
    int fidx     = (((w * 8 + cbk) * 32) + lane_s) * 16 + elem;
    ahi[fidx] = h;
    alo[fidx] = l;
  }
  __syncthreads();

  // ---- Phase 2: WMMA distance GEMM with fused running argmin ----
  const int wv = tid >> 5, lane = tid & 31;
  const int col = lane & 15, sel2 = lane >> 4;

  v16bf a_hi[8], a_lo[8];
  {
    const char* abase = smem + (size_t)(wv * 8 * 32 + lane) * 32;
    #pragma unroll
    for (int cbk = 0; cbk < 8; ++cbk) {
      a_hi[cbk] = __builtin_bit_cast(v16bf, *(const v8u*)(abase + cbk * 1024));
      a_lo[cbk] = __builtin_bit_cast(v16bf, *(const v8u*)(abase + OFF_ALO + cbk * 1024));
    }
  }

  float mv[8];
  int   mi[8];
  #pragma unroll
  for (int i = 0; i < 8; ++i) { mv[i] = 3.4e38f; mi[i] = 0; }

  // per-lane B base: code = kt*16+col, c-offset = sel2*16 (B 32x16 mapping)
  const char* bptr0 = (const char*)cb_hi + (size_t)(col * CDIM + sel2 * 16) * 2;

  #pragma unroll 1
  for (int kt = 0; kt < 64; ++kt) {
    const char* bbase = bptr0 + (size_t)kt * 16 * CDIM * 2;
    float en = enorm[kt * 16 + col];
    v16bf bf[8];
    #pragma unroll
    for (int cbk = 0; cbk < 8; ++cbk)
      bf[cbk] = __builtin_bit_cast(v16bf, *(const v8u*)(bbase + cbk * 64));
    if (kt < 63) __builtin_prefetch(bbase + 16 * CDIM * 2, 0, 3);

    // Two independent accumulation chains (hi-split / lo-split) so the XDL
    // pipe can interleave them instead of one 16-deep serial chain.
    v8f acch = {}, accl = {};
    #pragma unroll
    for (int cbk = 0; cbk < 8; ++cbk) {
      acch = __builtin_amdgcn_wmma_f32_16x16x32_bf16(false, a_hi[cbk], false, bf[cbk],
                                                     (short)0, acch, false, false);
      accl = __builtin_amdgcn_wmma_f32_16x16x32_bf16(false, a_lo[cbk], false, bf[cbk],
                                                     (short)0, accl, false, false);
    }
    int code = kt * 16 + col;
    #pragma unroll
    for (int i = 0; i < 8; ++i) {                 // dist = ||e||^2 - 2 z.e
      float d = __builtin_fmaf(-2.0f, acch[i], __builtin_fmaf(-2.0f, accl[i], en));
      if (d < mv[i]) { mv[i] = d; mi[i] = code; } // VALU co-executes with WMMA
    }
  }

  // C/D layout: VGPR i -> row i (lanes 0-15) / row 8+i (lanes 16-31), col = lane&15
  #pragma unroll
  for (int i = 0; i < 8; ++i) {
    int row = wv * 16 + i + ((lane >> 4) << 3);
    redv[row * 16 + col] = mv[i];
    redi[row * 16 + col] = mi[i];
  }
  __syncthreads();

  if (tid < M_TILE) {
    float bv = redv[tid * 16];
    int   bi = redi[tid * 16];
    #pragma unroll
    for (int k = 1; k < 16; ++k) {
      float v = redv[tid * 16 + k];
      int  ii = redi[tid * 16 + k];
      if (v < bv || (v == bv && ii < bi)) { bv = v; bi = ii; }
    }
    idxs[tid] = bi;
    out_idx[(size_t)tile * M_TILE + tid] = (float)bi;
  }
  __syncthreads();

  // ---- Phase 3a: async-copy winning codebook rows (fp32) into LDS ----
  // GLOBAL_LOAD_ASYNC_TO_LDS_B128: IOFFSET is added to BOTH the global and
  // the LDS address, so one base pair + 32 immediate offsets copies 512 B.
  {
    int j = tid >> 1, half = tid & 1;
    int code = idxs[j];
    unsigned ldsbase = (unsigned)__builtin_amdgcn_groupstaticsize() +
                       (unsigned)(j * ZQ_PITCH + half * 128) * 4u;
    unsigned goff = (unsigned)(code * CDIM + half * 128) * 4u;
    #pragma unroll
    for (int i = 0; i < 32; ++i) {
      asm volatile("global_load_async_to_lds_b128 %0, %1, %2 offset:%3"
                   :: "v"(ldsbase), "v"(goff), "s"(cb_f32), "n"(i * 16)
                   : "memory");
    }
    asm volatile("s_wait_asynccnt 0" ::: "memory");
  }
  __syncthreads();

  // ---- Phase 3b: coalesced z_q write + fused loss accumulation ----
  float lsum = 0.f;
  #pragma unroll 4
  for (int e = tid; e < M_TILE * CDIM; e += 256) {
    int c = e >> 7, j = e & (M_TILE - 1);
    float q = zq[j * ZQ_PITCH + c];
    size_t g = zbase + (size_t)c * HW + j;
    float ze = z_e[g];
    out_zq[g] = q;
    float d = q - ze;
    lsum += d * d;
  }
  lred[tid] = lsum;
  __syncthreads();
  for (int s = 128; s > 0; s >>= 1) {
    if (tid < s) lred[tid] += lred[tid + s];
    __syncthreads();
  }
  if (tid == 0) partials[tile] = lred[0];
}

// ---------------------------------------------------------------------------
// Kernel 2: deterministic fixed-order final loss reduction.
// ---------------------------------------------------------------------------
__global__ void vq_finalize(const float* __restrict__ partials,
                            float* __restrict__ out_loss) {
  __shared__ float s[256];
  int t = threadIdx.x;
  s[t] = partials[t] + partials[t + 256] + partials[t + 512] + partials[t + 768];
  __syncthreads();
  for (int k = 128; k > 0; k >>= 1) {
    if (t < k) s[t] += s[t + k];
    __syncthreads();
  }
  if (t == 0) *out_loss = s[0] * (1.25f / (float)ZQ_NUMEL);
}

// ---------------------------------------------------------------------------
extern "C" void kernel_launch(void* const* d_in, const int* in_sizes, int n_in,
                              void* d_out, int out_size, void* d_ws, size_t ws_size,
                              hipStream_t stream) {
  (void)in_sizes; (void)n_in; (void)out_size; (void)ws_size;
  const float* z_e = (const float*)d_in[0];
  const float* cbf = (const float*)d_in[1];

  float* out      = (float*)d_out;
  float* out_zq   = out;                       // [32,256,64,64]
  float* out_loss = out + ZQ_NUMEL;            // scalar
  float* out_idx  = out + ZQ_NUMEL + 1;        // [131072] indices (as float)

  unsigned short* cb_hi = (unsigned short*)d_ws;                          // 512 KB
  float* enorm    = (float*)((char*)d_ws + (size_t)NUM_CODES * CDIM * 2); // 4 KB
  float* partials = enorm + NUM_CODES;                                    // 4 KB

  vq_prep<<<NUM_CODES, 256, 0, stream>>>(cbf, cb_hi, enorm);
  vq_main<<<NTILES, 256, LDS_BYTES, stream>>>(z_e, cbf, cb_hi, enorm,
                                              out_zq, out_idx, partials);
  vq_finalize<<<1, 256, 0, stream>>>(partials, out_loss);
}
